// LocalAttention_55018531062339
// MI455X (gfx1250) — matrix operand order repaired and checked by reading, not compiled, not run on hardware
//
#include <hip/hip_runtime.h>
#include <math.h>

typedef __attribute__((ext_vector_type(16))) __bf16 v16bf;
typedef __attribute__((ext_vector_type(8)))  __bf16 v8bf;
typedef __attribute__((ext_vector_type(8)))  float  v8f;

#define S_LEN   2048
#define HEADS   16
#define HDIM    64
#define EMB     1024
#define BATCH   2
#define SPAD    2368        // S + 128 (front pad) + 192 (back pad)
#define POFF    128
#define QTILES  128         // S/16

// ---------------------------------------------------------------------------
// Zero-fill helpers (attn output region must be exactly 0 outside the window;
// Vt pads must be finite so pad*prob(0) == 0).
// ---------------------------------------------------------------------------
__global__ void zero_f4(float4* __restrict__ p, size_t n) {
    size_t i = (size_t)blockIdx.x * blockDim.x + threadIdx.x;
    size_t stride = (size_t)gridDim.x * blockDim.x;
    float4 z; z.x = 0.f; z.y = 0.f; z.z = 0.f; z.w = 0.f;
    for (; i < n; i += stride) p[i] = z;
}

__global__ void zero_u32(unsigned* __restrict__ p, size_t n) {
    size_t i = (size_t)blockIdx.x * blockDim.x + threadIdx.x;
    size_t stride = (size_t)gridDim.x * blockDim.x;
    for (; i < n; i += stride) p[i] = 0u;
}

// One-time fp32 -> bf16 streaming conversion (halves all repeated GEMM reads)
__global__ void cvt_bf16(const float* __restrict__ s, __bf16* __restrict__ d, size_t n) {
    size_t i = ((size_t)blockIdx.x * blockDim.x + threadIdx.x) * 8;
    size_t stride = (size_t)gridDim.x * blockDim.x * 8;
    for (; i < n; i += stride) {
        float4 a = *(const float4*)(s + i);
        float4 b = *(const float4*)(s + i + 4);
        v8bf o;
        o[0] = (__bf16)a.x; o[1] = (__bf16)a.y; o[2] = (__bf16)a.z; o[3] = (__bf16)a.w;
        o[4] = (__bf16)b.x; o[5] = (__bf16)b.y; o[6] = (__bf16)b.z; o[7] = (__bf16)b.w;
        *(v8bf*)(d + i) = o;
    }
}

// ---------------------------------------------------------------------------
// Unified projection GEMM (all-bf16 operands): Y[m,f] = X[m,:] . W[f,:] + bias[f]
// One wave computes a 16(M) x 64(N) tile; K-loop of 32 with 4 WMMAs/iter.
// mode 0: Q  -> bf16 [b,h,s,d]
// mode 1: K  -> bf16 [b,h,s+POFF (SPAD rows),d]   (row padded, branchless tiles)
// mode 2: V  -> bf16 transposed [b,h,d,s+POFF (SPAD cols)]
// mode 3: O  -> fp32 [m, f] (final output)
// ---------------------------------------------------------------------------
__global__ __launch_bounds__(128)
void gemm_proj(const __bf16* __restrict__ X, const __bf16* __restrict__ W,
               const float* __restrict__ bias, __bf16* __restrict__ Ybf,
               float* __restrict__ Yf, int mode)
{
    const int lane  = threadIdx.x & 31;
    const int wv    = threadIdx.x >> 5;
    const int lm    = lane & 15;     // A row / B,C column within tile
    const int half  = lane >> 4;     // lane-half selects K/M sub-ranges
    const int mtile = blockIdx.x;                       // 0..255
    const int nbase = (blockIdx.y * 4 + wv) * 64;       // 0..960

    v8f acc[4] = {};

    const __bf16* xrow = X + (size_t)(mtile * 16 + lm) * EMB;

    for (int kk = 0; kk < EMB; kk += 32) {
        // A matrix: 16x32 bf16, contiguous order = 16 contiguous K per lane half
        const __bf16* p0 = xrow + kk + 16 * half;
        __builtin_prefetch(xrow + kk + 64, 0, 1);
        v16bf a;
#pragma unroll
        for (int j = 0; j < 8; ++j) {
            a[j]     = p0[j];
            a[j + 8] = p0[8 + j];
        }
#pragma unroll
        for (int nt = 0; nt < 4; ++nt) {
            // B matrix: 32x16 bf16; lane = column (= W row), 16 contiguous K (32B)
            const __bf16* wp = W + (size_t)(nbase + nt * 16 + lm) * EMB + kk + 16 * half;
            v16bf bm;
#pragma unroll
            for (int j = 0; j < 16; ++j) bm[j] = wp[j];
            acc[nt] = __builtin_amdgcn_wmma_f32_16x16x32_bf16(
                false, a, false, bm, (short)0, acc[nt], false, false);
        }
    }

#pragma unroll
    for (int nt = 0; nt < 4; ++nt) {
        const int f  = nbase + nt * 16 + lm;
        const float bv = bias[f];
        const int h = f >> 6, d = f & 63;
#pragma unroll
        for (int r = 0; r < 8; ++r) {
            const int mrow = mtile * 16 + r + 8 * half;
            const int b = mrow >> 11;            // /S_LEN
            const int s = mrow & (S_LEN - 1);
            const float y = acc[nt][r] + bv;
            if (mode == 0)
                Ybf[(((size_t)(b * HEADS + h)) * S_LEN + s) * HDIM + d] = (__bf16)y;
            else if (mode == 1)
                Ybf[(((size_t)(b * HEADS + h)) * SPAD + (s + POFF)) * HDIM + d] = (__bf16)y;
            else if (mode == 2)
                Ybf[(((size_t)(b * HEADS + h)) * HDIM + d) * SPAD + (s + POFF)] = (__bf16)y;
            else
                Yf[(size_t)mrow * EMB + f] = y;
        }
    }
}

// ---------------------------------------------------------------------------
// Local attention: one wave per (b,h, 16-query tile); 17 key tiles, branchless
// via K-row padding; masked cndmask kills pad garbage (incl. NaN).
// ---------------------------------------------------------------------------
__global__ __launch_bounds__(128)
void attn_kernel(const __bf16* __restrict__ Q, const __bf16* __restrict__ Kp,
                 const __bf16* __restrict__ Vt, float* __restrict__ attn,
                 __bf16* __restrict__ Oi)
{
    __shared__ __bf16 plds[4][16][288];   // per-wave bf16 probs, 18 key tiles

    const int lane = threadIdx.x & 31;
    const int wv   = threadIdx.x >> 5;
    const int lm   = lane & 15;
    const int half = lane >> 4;
    const int wid  = blockIdx.x * 4 + wv;       // 0..4095
    const int qt   = wid & (QTILES - 1);
    const int h    = (wid >> 7) & (HEADS - 1);
    const int b    = wid >> 11;
    const int bh   = b * HEADS + h;

    // Load Q tile A-registers (K-dim = 64 -> two K=32 chunks)
    const __bf16* qrow = Q + (((size_t)bh) * S_LEN + qt * 16 + lm) * HDIM;
    v16bf aq0, aq1;
#pragma unroll
    for (int j = 0; j < 8; ++j) {
        aq0[j]     = qrow[16 * half + j];
        aq0[j + 8] = qrow[8 + 16 * half + j];
        aq1[j]     = qrow[32 + 16 * half + j];
        aq1[j + 8] = qrow[40 + 16 * half + j];
    }

    v8f  sc[17];
    float rm[8], rs[8];
#pragma unroll
    for (int r = 0; r < 8; ++r) { rm[r] = -3.0e38f; rs[r] = 0.f; }

    const __bf16* kbase = Kp + ((size_t)bh) * SPAD * HDIM;
#pragma unroll
    for (int i = 0; i < 17; ++i) {
        const int srow = (qt - 8 + i) * 16 + lm + POFF;   // always in [0,SPAD)
        const __bf16* kr = kbase + (size_t)srow * HDIM;
        v16bf b0, b1;
#pragma unroll
        for (int j = 0; j < 16; ++j) {
            b0[j] = kr[16 * half + j];
            b1[j] = kr[32 + 16 * half + j];
        }
        v8f c = {};
        c = __builtin_amdgcn_wmma_f32_16x16x32_bf16(false, aq0, false, b0, (short)0, c, false, false);
        c = __builtin_amdgcn_wmma_f32_16x16x32_bf16(false, aq1, false, b1, (short)0, c, false, false);

        const int k = (qt - 8 + i) * 16 + lm;
#pragma unroll
        for (int r = 0; r < 8; ++r) {
            const int q  = qt * 16 + r + 8 * half;
            int dd = q - k; if (dd < 0) dd = -dd;
            const bool valid = (k >= 0) && (k < S_LEN) && (dd <= 128);
            const float s = valid ? c[r] * 0.125f : -1.0e9f;   // 1/sqrt(64)
            sc[i][r] = s;
            rm[r] = fmaxf(rm[r], s);
        }
    }

    // Row max across the 16 lanes of each half-group
#pragma unroll
    for (int r = 0; r < 8; ++r) {
        float v = rm[r];
        v = fmaxf(v, __shfl_xor(v, 1));
        v = fmaxf(v, __shfl_xor(v, 2));
        v = fmaxf(v, __shfl_xor(v, 4));
        v = fmaxf(v, __shfl_xor(v, 8));
        rm[r] = v;
    }
#pragma unroll
    for (int i = 0; i < 17; ++i)
#pragma unroll
        for (int r = 0; r < 8; ++r) {
            const float p = __expf(sc[i][r] - rm[r]);
            sc[i][r] = p;
            rs[r] += p;
        }
#pragma unroll
    for (int r = 0; r < 8; ++r) {
        float v = rs[r];
        v += __shfl_xor(v, 1); v += __shfl_xor(v, 2);
        v += __shfl_xor(v, 4); v += __shfl_xor(v, 8);
        rs[r] = 1.0f / v;   // diagonal element guarantees sum >= 1
    }

    // Emit fp32 probs to global attn + bf16 to LDS (A-side staging for PV)
    float* abase = attn + ((size_t)bh) * S_LEN * S_LEN;
#pragma unroll
    for (int i = 0; i < 17; ++i) {
        const int k = (qt - 8 + i) * 16 + lm;
        const bool kin = (k >= 0) && (k < S_LEN);
#pragma unroll
        for (int r = 0; r < 8; ++r) {
            const int m = r + 8 * half;
            const float pr = sc[i][r] * rs[r];
            if (kin) abase[(size_t)(qt * 16 + m) * S_LEN + k] = pr;
            plds[wv][m][i * 16 + lm] = (__bf16)pr;
        }
    }
    // Zero the 18th (padding) tile so the upper 16 keys of the 9th K=32 chunk contribute nothing
#pragma unroll
    for (int r = 0; r < 8; ++r) plds[wv][r + 8 * half][272 + lm] = (__bf16)0.0f;

    // P @ V : M=16 queries, N=64 dims, K=288 keys (9 chunks of 32)
    v8f o[4] = {};
    const __bf16* vb = Vt + ((size_t)bh) * HDIM * SPAD;
#pragma unroll
    for (int ch = 0; ch < 9; ++ch) {
        v16bf a;
#pragma unroll
        for (int j = 0; j < 8; ++j) {
            a[j]     = plds[wv][lm][ch * 32 + 16 * half + j];
            a[j + 8] = plds[wv][lm][ch * 32 + 8 + 16 * half + j];
        }
#pragma unroll
        for (int nt = 0; nt < 4; ++nt) {
            // Vt column run: (qt-8)*16 + POFF == qt*16, contiguous 16 keys (32B)
            const __bf16* vr = vb + (size_t)(nt * 16 + lm) * SPAD
                                  + qt * 16 + ch * 32 + 16 * half;
            v16bf bm;
#pragma unroll
            for (int j = 0; j < 16; ++j) bm[j] = vr[j];
            o[nt] = __builtin_amdgcn_wmma_f32_16x16x32_bf16(
                false, a, false, bm, (short)0, o[nt], false, false);
        }
    }

    // Store O tile directly as bf16 [m, e] — it only feeds the final bf16 GEMM
    __bf16* orow = Oi + ((size_t)(b * S_LEN + qt * 16)) * EMB + h * HDIM;
#pragma unroll
    for (int nt = 0; nt < 4; ++nt)
#pragma unroll
        for (int r = 0; r < 8; ++r)
            orow[(size_t)(r + 8 * half) * EMB + nt * 16 + lm] = (__bf16)o[nt][r];
}

// ---------------------------------------------------------------------------
extern "C" void kernel_launch(void* const* d_in, const int* in_sizes, int n_in,
                              void* d_out, int out_size, void* d_ws, size_t ws_size,
                              hipStream_t stream)
{
    const float* x  = (const float*)d_in[0];
    const float* Wq = (const float*)d_in[1];
    const float* bq = (const float*)d_in[2];
    const float* Wk = (const float*)d_in[3];
    const float* bk = (const float*)d_in[4];
    const float* Wv = (const float*)d_in[5];
    const float* bv = (const float*)d_in[6];
    const float* Wo = (const float*)d_in[7];
    const float* bo = (const float*)d_in[8];

    float* out  = (float*)d_out;                               // [B,S,E]
    float* attn = out + (size_t)BATCH * S_LEN * EMB;           // [B,H,S,S]

    // Workspace layout (bytes):
    //   xb  bf16 [B*S, E]            :  8,388,608
    //   Wqb/Wkb/Wvb/Wob bf16 [E,E]   :  4 x 2,097,152
    //   Q   bf16 [B,H,S,64]          :  8,388,608
    //   Kp  bf16 [B,H,SPAD,64]       :  9,699,328
    //   Vt  bf16 [B,H,64,SPAD]       :  9,699,328
    //   Oi  bf16 [B*S, E]            :  8,388,608     (total ~50.5 MB)
    char* ws = (char*)d_ws;
    size_t off = 0;
    __bf16* xb  = (__bf16*)(ws + off); off += 8388608;
    __bf16* Wqb = (__bf16*)(ws + off); off += 2097152;
    __bf16* Wkb = (__bf16*)(ws + off); off += 2097152;
    __bf16* Wvb = (__bf16*)(ws + off); off += 2097152;
    __bf16* Wob = (__bf16*)(ws + off); off += 2097152;
    __bf16* Qb  = (__bf16*)(ws + off); off += 8388608;
    __bf16* Kb  = (__bf16*)(ws + off); off += 9699328;
    __bf16* Vtb = (__bf16*)(ws + off); off += 9699328;
    __bf16* Oib = (__bf16*)(ws + off); off += 8388608;

    // 1) Zero the attn output (exact zeros outside window) and Vt (finite pads)
    zero_f4 <<<2048, 256, 0, stream>>>((float4*)attn,
                                       (size_t)BATCH * HEADS * S_LEN * S_LEN / 4);
    zero_u32<<<512,  256, 0, stream>>>((unsigned*)Vtb,
                                       (size_t)BATCH * HEADS * HDIM * SPAD / 2);

    // 2) One-time bf16 conversion of activations + weights
    cvt_bf16<<<1024, 256, 0, stream>>>(x,  xb,  (size_t)BATCH * S_LEN * EMB);
    cvt_bf16<<<512,  256, 0, stream>>>(Wq, Wqb, (size_t)EMB * EMB);
    cvt_bf16<<<512,  256, 0, stream>>>(Wk, Wkb, (size_t)EMB * EMB);
    cvt_bf16<<<512,  256, 0, stream>>>(Wv, Wvb, (size_t)EMB * EMB);
    cvt_bf16<<<512,  256, 0, stream>>>(Wo, Wob, (size_t)EMB * EMB);

    // 3) Q/K/V projections (pure-bf16 WMMA GEMMs)
    dim3 g(256, 4, 1);
    gemm_proj<<<g, 128, 0, stream>>>(xb, Wqb, bq, Qb,  nullptr, 0);
    gemm_proj<<<g, 128, 0, stream>>>(xb, Wkb, bk, Kb,  nullptr, 1);
    gemm_proj<<<g, 128, 0, stream>>>(xb, Wvb, bv, Vtb, nullptr, 2);

    // 4) Windowed attention (scores WMMA -> softmax -> attn write -> PV WMMA)
    attn_kernel<<<1024, 128, 0, stream>>>(Qb, Kb, Vtb, attn, Oib);

    // 5) Output projection into d_out
    gemm_proj<<<g, 128, 0, stream>>>(Oib, Wob, bo, nullptr, out, 3);
}
